// EGNNDipoleModel_66511863545965
// MI455X (gfx1250) — matrix operand order repaired and checked by reading, not compiled, not run on hardware
//
#include <hip/hip_runtime.h>

#define NN 50000
#define NE 1600000
#define NB 512
#define NF 64
#define NH 64
#define NLAY 4
#define EDGE_IN 129
#define W1T_STRIDE 144  // 129 padded to 16-element multiple (32B rows)

typedef __attribute__((ext_vector_type(16))) __bf16 v16bf;
typedef __attribute__((ext_vector_type(8)))  __bf16 v8bf;
typedef __attribute__((ext_vector_type(8)))  float  v8f;

__device__ __forceinline__ __bf16 f2bf(float f) {
  unsigned u = __builtin_bit_cast(unsigned, f);
  unsigned short hs = (unsigned short)(u >> 16);
  return __builtin_bit_cast(__bf16, hs);
}
__device__ __forceinline__ float bf2f(__bf16 b) {
  unsigned short hs = __builtin_bit_cast(unsigned short, b);
  unsigned u = ((unsigned)hs) << 16;
  return __builtin_bit_cast(float, u);
}
__device__ __forceinline__ float silu_f(float v) {
  return v * (1.0f / (1.0f + __expf(-v)));
}
// Intra-wave LDS producer->consumer fence (cross-lane through LDS within one
// wave): CDNA5 split counter wait.
__device__ __forceinline__ void lds_wave_fence() {
  asm volatile("s_wait_dscnt 0x0" ::: "memory");
}
// Async global->LDS 16B copy (CDNA5, tracked with ASYNCcnt). LDS destination
// offset = addr[31:0] of the generic pointer (LDS aperture rule, ISA 10.2).
__device__ __forceinline__ void async_ld_b128(const __bf16* gsrc, __bf16* ldst) {
  const unsigned loff = (unsigned)(uintptr_t)ldst;
  asm volatile("global_load_async_to_lds_b128 %0, %1, off"
               :: "v"(loff), "v"(gsrc) : "memory");
}
__device__ __forceinline__ void wait_async0() {
  asm volatile("s_wait_asynccnt 0x0" ::: "memory");
}
__device__ __forceinline__ v16bf cat8(v8bf lo, v8bf hi) {
  return __builtin_shufflevector(lo, hi, 0, 1, 2, 3, 4, 5, 6, 7,
                                 8, 9, 10, 11, 12, 13, 14, 15);
}

// ---- WMMA fragment helpers (wave32, v_wmma_f32_16x16x32_bf16) -------------
// B fragment from TRANSPOSED LDS weights Wt[n][k] (row stride `stride`, 32B
// aligned rows). Lane (n = (lane&15)+nbase, half) holds K = kbase +
// {8*half..+7} and {16+8*half..+7} -> two ds_load_b128.
__device__ __forceinline__ v16bf load_b_frag_t(const __bf16* Wt, int stride,
                                               int kbase, int nbase, int lane) {
  const int half = lane >> 4;
  const int n = (lane & 15) + nbase;
  const __bf16* p = Wt + n * stride + kbase + 8 * half;
  return cat8(*(const v8bf*)(p), *(const v8bf*)(p + 16));
}
// A fragment from bf16 row data (global or LDS): lane (m=lane&15, half) holds
// k = {8*half..+7} and {16+8*half..+7} relative to rowp -> two b128 loads.
__device__ __forceinline__ v16bf load_a_frag_bf16(const __bf16* rowp, int lane) {
  const int half = lane >> 4;
  return cat8(*(const v8bf*)(rowp + 8 * half),
              *(const v8bf*)(rowp + 16 + 8 * half));
}
// A fragment from f32 row data (m_agg): two b128 load pairs + truncate-cvt.
__device__ __forceinline__ v16bf load_a_frag_f32(const float* rowp, int lane) {
  const int half = lane >> 4;
  v16bf a;
#pragma unroll
  for (int e = 0; e < 8; ++e) a[e] = f2bf(rowp[8 * half + e]);
#pragma unroll
  for (int e = 0; e < 8; ++e) a[8 + e] = f2bf(rowp[16 + 8 * half + e]);
  return a;
}
#define WMMA_BF16(A, B, C) \
  __builtin_amdgcn_wmma_f32_16x16x32_bf16(false, (A), false, (B), (short)0, (C), false, false)

// ---- prep: convert all weights to bf16 and transpose to [n][k] once --------
__global__ __launch_bounds__(256) void k_prep(
    const float* __restrict__ eW1, const float* __restrict__ eW2,
    const float* __restrict__ nW1, const float* __restrict__ nW2,
    const float* __restrict__ qW1,
    __bf16* __restrict__ eW1t, __bf16* __restrict__ eW2t,
    __bf16* __restrict__ nW1t, __bf16* __restrict__ nW2t,
    __bf16* __restrict__ qW1t) {
  const int t = blockIdx.x * 256 + threadIdx.x;
  if (t < NLAY * EDGE_IN * 64) {  // [l][k][n] -> [l][n][W1T_STRIDE]+k
    const int l = t / (EDGE_IN * 64), rem = t % (EDGE_IN * 64);
    const int k = rem >> 6, n = rem & 63;
    eW1t[(l * 64 + n) * W1T_STRIDE + k] = f2bf(eW1[t]);
  }
  if (t < NLAY * 64 * 64) {
    const int l = t >> 12, rem = t & 4095;
    const int k = rem >> 6, n = rem & 63;
    eW2t[(l * 64 + n) * 64 + k] = f2bf(eW2[t]);
    nW2t[(l * 64 + n) * 64 + k] = f2bf(nW2[t]);
  }
  if (t < NLAY * 128 * 64) {
    const int l = t >> 13, rem = t & 8191;
    const int k = rem >> 6, n = rem & 63;
    nW1t[(l * 64 + n) * 128 + k] = f2bf(nW1[t]);
  }
  if (t < 64 * 64) {
    const int k = t >> 6, n = t & 63;
    qW1t[n * 64 + k] = f2bf(qW1[t]);
  }
}

// ---- init: h = emb[z] (bf16), x = pos, zero accumulators -------------------
__global__ __launch_bounds__(256) void k_init(
    const int* __restrict__ z, const float* __restrict__ pos, const float* __restrict__ emb,
    __bf16* __restrict__ hb, float* __restrict__ x, float* __restrict__ m_agg,
    float* __restrict__ dxb, float* __restrict__ cnt, float* __restrict__ xsum) {
  const int t = blockIdx.x * 256 + threadIdx.x;
  if (t < NN * NF) {
    const int n = t >> 6, f = t & 63;
    hb[t] = f2bf(emb[z[n] * NF + f]);
    m_agg[t] = 0.0f;
  }
  if (t < NN * 3) { x[t] = pos[t]; dxb[t] = 0.0f; }
  if (t < NB)     cnt[t] = 0.0f;
  if (t < NB * 3) xsum[t] = 0.0f;
}

// ---- edge MLP: m = silu(silu([h_r,h_c,d2]@W1+b1)@W2+b2), coef = tanh(m@cW+cb)
// scatter-add m into m_agg[row], rel*coef into dx[row]. One wave = 16 edges.
__global__ __launch_bounds__(256) void k_edge(
    const __bf16* __restrict__ hb, const float* __restrict__ x,
    float* __restrict__ m_agg, float* __restrict__ dxb,
    const int* __restrict__ erow, const int* __restrict__ ecol,
    const __bf16* __restrict__ W1t, const float* __restrict__ b1,
    const __bf16* __restrict__ W2t, const float* __restrict__ b2,
    const float* __restrict__ cWp, const float* __restrict__ cbp) {
  __shared__ __align__(32) __bf16 w1t[64 * W1T_STRIDE];  // transposed [n][k]
  __shared__ __align__(32) __bf16 w2t[64 * 64];          // transposed [n][k]
  __shared__ float sb1[64], sb2[64], scw[64];
  __shared__ __align__(16) float relb[8][48];
  __shared__ __align__(16) float d2b[8][16];
  __shared__ int   rowb[8][16];
  __shared__ float coefb[8][16];
  __shared__ __align__(32) __bf16 stage[8][16 * 64];

  const int t = threadIdx.x;
  // async-copy pre-transposed bf16 weights into LDS (16B chunks, ASYNCcnt)
  for (int i = t; i < (64 * W1T_STRIDE) / 8; i += 256)
    async_ld_b128(W1t + i * 8, &w1t[i * 8]);
  for (int i = t; i < (64 * 64) / 8; i += 256)
    async_ld_b128(W2t + i * 8, &w2t[i * 8]);
  if (t < 64) { sb1[t] = b1[t]; sb2[t] = b2[t]; scw[t] = cWp[t]; }
  wait_async0();
  __syncthreads();
  const float cbv = cbp[0];

  const int wave = t >> 5, lane = t & 31;
  const int half = lane >> 4, mi = lane & 15;
  const int e = (blockIdx.x * 8 + wave) * 16 + mi;  // grid divides E exactly
  const int r = erow[e], c = ecol[e];
  const float rx = x[r * 3 + 0] - x[c * 3 + 0];
  const float ry = x[r * 3 + 1] - x[c * 3 + 1];
  const float rz = x[r * 3 + 2] - x[c * 3 + 2];
  const float d2 = rx * rx + ry * ry + rz * rz;
  if (half == 0) {
    relb[wave][mi * 3 + 0] = rx; relb[wave][mi * 3 + 1] = ry; relb[wave][mi * 3 + 2] = rz;
    d2b[wave][mi] = d2; rowb[wave][mi] = r;
  }
  lds_wave_fence();

  // Hoisted accumulator-init data: dist2 of this lane's 8 C-rows (vectorized)
  const float4 d2lo = *(const float4*)&d2b[wave][8 * half];
  const float4 d2hi = *(const float4*)&d2b[wave][8 * half + 4];
  const float d2v[8] = {d2lo.x, d2lo.y, d2lo.z, d2lo.w, d2hi.x, d2hi.y, d2hi.z, d2hi.w};

  // A fragments: gathered bf16 node features (K=64 each, 2 chunks of 32)
  const v16bf aR0 = load_a_frag_bf16(hb + r * 64 + 0,  lane);
  const v16bf aR1 = load_a_frag_bf16(hb + r * 64 + 32, lane);
  const v16bf aC0 = load_a_frag_bf16(hb + c * 64 + 0,  lane);
  const v16bf aC1 = load_a_frag_bf16(hb + c * 64 + 32, lane);

  // GEMM1: [16,129]@[129,64]; dist2 rank-1 term + bias folded into acc init
#pragma unroll
  for (int nt = 0; nt < 4; ++nt) {
    const int n = mi + nt * 16;
    const float w128n = bf2f(w1t[n * W1T_STRIDE + 128]);
    v8f acc;
#pragma unroll
    for (int j = 0; j < 8; ++j) acc[j] = sb1[n] + d2v[j] * w128n;
    acc = WMMA_BF16(aR0, load_b_frag_t(w1t, W1T_STRIDE,  0, nt * 16, lane), acc);
    acc = WMMA_BF16(aR1, load_b_frag_t(w1t, W1T_STRIDE, 32, nt * 16, lane), acc);
    acc = WMMA_BF16(aC0, load_b_frag_t(w1t, W1T_STRIDE, 64, nt * 16, lane), acc);
    acc = WMMA_BF16(aC1, load_b_frag_t(w1t, W1T_STRIDE, 96, nt * 16, lane), acc);
#pragma unroll
    for (int j = 0; j < 8; ++j)
      stage[wave][(j + 8 * half) * 64 + n] = f2bf(silu_f(acc[j]));  // C->A transpose via LDS
  }
  lds_wave_fence();

  const v16bf g0 = load_a_frag_bf16(&stage[wave][mi * 64 + 0],  lane);
  const v16bf g1 = load_a_frag_bf16(&stage[wave][mi * 64 + 32], lane);

  float msg[4][8];
  float part[8];
#pragma unroll
  for (int j = 0; j < 8; ++j) part[j] = 0.0f;

  // GEMM2: [16,64]@[64,64] + silu; accumulate coef partials m*cW on the fly
#pragma unroll
  for (int nt = 0; nt < 4; ++nt) {
    const int n = mi + nt * 16;
    v8f acc;
#pragma unroll
    for (int j = 0; j < 8; ++j) acc[j] = sb2[n];
    acc = WMMA_BF16(g0, load_b_frag_t(w2t, 64,  0, nt * 16, lane), acc);
    acc = WMMA_BF16(g1, load_b_frag_t(w2t, 64, 32, nt * 16, lane), acc);
    const float cwn = scw[n];
#pragma unroll
    for (int j = 0; j < 8; ++j) {
      const float mv = silu_f(acc[j]);
      msg[nt][j] = mv;
      part[j] += mv * cwn;
    }
  }

  // reduce over the 16 lanes of each half (wave32, xor bits 0..3 stay in half)
#pragma unroll
  for (int j = 0; j < 8; ++j) {
    float p = part[j];
    p += __shfl_xor(p, 1, 32);
    p += __shfl_xor(p, 2, 32);
    p += __shfl_xor(p, 4, 32);
    p += __shfl_xor(p, 8, 32);
    const float cf = tanhf(p + cbv);
    if (mi == j) coefb[wave][j + 8 * half] = cf;
  }

  // scatter m into m_agg[row] (f32 atomics: full-precision segment sum)
#pragma unroll
  for (int j = 0; j < 8; ++j) {
    const int rr = rowb[wave][j + 8 * half];
#pragma unroll
    for (int nt = 0; nt < 4; ++nt)
      atomicAdd(&m_agg[rr * 64 + mi + nt * 16], msg[nt][j]);
  }

  lds_wave_fence();
  if (half == 0) {
    const float cf = coefb[wave][mi];
    const int rr = rowb[wave][mi];
    atomicAdd(&dxb[rr * 3 + 0], relb[wave][mi * 3 + 0] * cf);
    atomicAdd(&dxb[rr * 3 + 1], relb[wave][mi * 3 + 1] * cf);
    atomicAdd(&dxb[rr * 3 + 2], relb[wave][mi * 3 + 2] * cf);
  }
}

// ---- node MLP: x += dx; h += silu([h,m_agg]@W1+b1)@W2+b2; re-zero accums ---
__global__ __launch_bounds__(256) void k_node(
    __bf16* __restrict__ hb, float* __restrict__ x,
    float* __restrict__ m_agg, float* __restrict__ dxb,
    const __bf16* __restrict__ W1t, const float* __restrict__ b1,
    const __bf16* __restrict__ W2t, const float* __restrict__ b2) {
  __shared__ __align__(32) __bf16 w1t[64 * 128];  // transposed [n][k]
  __shared__ __align__(32) __bf16 w2t[64 * 64];   // transposed [n][k]
  __shared__ float sb1[64], sb2[64];
  __shared__ __align__(32) __bf16 stage[8][16 * 64];

  const int t = threadIdx.x;
  for (int i = t; i < (64 * 128) / 8; i += 256)
    async_ld_b128(W1t + i * 8, &w1t[i * 8]);
  for (int i = t; i < (64 * 64) / 8; i += 256)
    async_ld_b128(W2t + i * 8, &w2t[i * 8]);
  if (t < 64) { sb1[t] = b1[t]; sb2[t] = b2[t]; }
  wait_async0();
  __syncthreads();

  const int wave = t >> 5, lane = t & 31;
  const int half = lane >> 4, mi = lane & 15;
  const int tile = blockIdx.x * 8 + wave;
  if (tile * 16 >= NN) return;  // wave-uniform; no barriers below
  const int node = tile * 16 + mi;

  if (half == 0) {
#pragma unroll
    for (int k = 0; k < 3; ++k) {
      x[node * 3 + k] += dxb[node * 3 + k];
      dxb[node * 3 + k] = 0.0f;  // ready for next layer
    }
  }

  const v16bf a0 = load_a_frag_bf16(hb    + node * 64 + 0,  lane);
  const v16bf a1 = load_a_frag_bf16(hb    + node * 64 + 32, lane);
  const v16bf a2 = load_a_frag_f32(m_agg  + node * 64 + 0,  lane);
  const v16bf a3 = load_a_frag_f32(m_agg  + node * 64 + 32, lane);

#pragma unroll
  for (int nt = 0; nt < 4; ++nt) {
    const int n = mi + nt * 16;
    v8f acc;
#pragma unroll
    for (int j = 0; j < 8; ++j) acc[j] = sb1[n];
    acc = WMMA_BF16(a0, load_b_frag_t(w1t, 128,  0, nt * 16, lane), acc);
    acc = WMMA_BF16(a1, load_b_frag_t(w1t, 128, 32, nt * 16, lane), acc);
    acc = WMMA_BF16(a2, load_b_frag_t(w1t, 128, 64, nt * 16, lane), acc);
    acc = WMMA_BF16(a3, load_b_frag_t(w1t, 128, 96, nt * 16, lane), acc);
#pragma unroll
    for (int j = 0; j < 8; ++j)
      stage[wave][(j + 8 * half) * 64 + n] = f2bf(silu_f(acc[j]));
  }
  lds_wave_fence();

  const v16bf g0 = load_a_frag_bf16(&stage[wave][mi * 64 + 0],  lane);
  const v16bf g1 = load_a_frag_bf16(&stage[wave][mi * 64 + 32], lane);

#pragma unroll
  for (int nt = 0; nt < 4; ++nt) {
    const int n = mi + nt * 16;
    v8f acc;
#pragma unroll
    for (int j = 0; j < 8; ++j) acc[j] = sb2[n];
    acc = WMMA_BF16(g0, load_b_frag_t(w2t, 64,  0, nt * 16, lane), acc);
    acc = WMMA_BF16(g1, load_b_frag_t(w2t, 64, 32, nt * 16, lane), acc);
#pragma unroll
    for (int j = 0; j < 8; ++j) {
      const int nn = tile * 16 + j + 8 * half;
      const float hv = bf2f(hb[nn * 64 + n]);
      hb[nn * 64 + n] = f2bf(hv + acc[j]);  // residual
      m_agg[nn * 64 + n] = 0.0f;            // ready for next layer
    }
  }
}

// ---- q head: q = silu(h@qW1+qb1)@qW2+qb2; accumulate per-graph cnt / xsum --
__global__ __launch_bounds__(256) void k_q(
    const __bf16* __restrict__ hb, const float* __restrict__ x, const int* __restrict__ batch,
    const __bf16* __restrict__ W1t, const float* __restrict__ b1,
    const float* __restrict__ W2v, const float* __restrict__ b2,
    float* __restrict__ q, float* __restrict__ cnt, float* __restrict__ xsum) {
  __shared__ __align__(32) __bf16 w1t[64 * 64];  // transposed [n][k]
  __shared__ float sb1[64], sw2[64];
  const int t = threadIdx.x;
  for (int i = t; i < (64 * 64) / 8; i += 256)
    async_ld_b128(W1t + i * 8, &w1t[i * 8]);
  if (t < 64) { sb1[t] = b1[t]; sw2[t] = W2v[t]; }
  wait_async0();
  __syncthreads();
  const float qb2v = b2[0];

  const int wave = t >> 5, lane = t & 31;
  const int half = lane >> 4, mi = lane & 15;
  const int tile = blockIdx.x * 8 + wave;
  if (tile * 16 >= NN) return;
  const int node = tile * 16 + mi;

  const v16bf a0 = load_a_frag_bf16(hb + node * 64 + 0,  lane);
  const v16bf a1 = load_a_frag_bf16(hb + node * 64 + 32, lane);

  float part[8];
#pragma unroll
  for (int j = 0; j < 8; ++j) part[j] = 0.0f;

#pragma unroll
  for (int nt = 0; nt < 4; ++nt) {
    const int n = mi + nt * 16;
    v8f acc;
#pragma unroll
    for (int j = 0; j < 8; ++j) acc[j] = sb1[n];
    acc = WMMA_BF16(a0, load_b_frag_t(w1t, 64,  0, nt * 16, lane), acc);
    acc = WMMA_BF16(a1, load_b_frag_t(w1t, 64, 32, nt * 16, lane), acc);
    const float wn = sw2[n];
#pragma unroll
    for (int j = 0; j < 8; ++j) part[j] += silu_f(acc[j]) * wn;
  }

#pragma unroll
  for (int j = 0; j < 8; ++j) {
    float p = part[j];
    p += __shfl_xor(p, 1, 32);
    p += __shfl_xor(p, 2, 32);
    p += __shfl_xor(p, 4, 32);
    p += __shfl_xor(p, 8, 32);
    if (mi == j) q[tile * 16 + j + 8 * half] = p + qb2v;
  }

  if (half == 0) {
    const int b = batch[node];
    atomicAdd(&cnt[b], 1.0f);
#pragma unroll
    for (int k = 0; k < 3; ++k) atomicAdd(&xsum[b * 3 + k], x[node * 3 + k]);
  }
}

// ---- centers; zero mu ------------------------------------------------------
__global__ __launch_bounds__(512) void k_center(
    const float* __restrict__ cnt, const float* __restrict__ xsum,
    float* __restrict__ center, float* __restrict__ mu) {
  const int b = threadIdx.x;
  const float c = fmaxf(cnt[b], 1.0f);
#pragma unroll
  for (int k = 0; k < 3; ++k) {
    center[b * 3 + k] = xsum[b * 3 + k] / c;
    mu[b * 3 + k] = 0.0f;
  }
}

// ---- mu = segment_sum(q * (x - center[batch])) -----------------------------
__global__ __launch_bounds__(256) void k_mu(
    const float* __restrict__ x, const float* __restrict__ q, const int* __restrict__ batch,
    const float* __restrict__ center, float* __restrict__ mu) {
  const int n = blockIdx.x * 256 + threadIdx.x;
  if (n < NN) {
    const int b = batch[n];
    const float qq = q[n];
#pragma unroll
    for (int k = 0; k < 3; ++k)
      atomicAdd(&mu[b * 3 + k], qq * (x[n * 3 + k] - center[b * 3 + k]));
  }
}

extern "C" void kernel_launch(void* const* d_in, const int* in_sizes, int n_in,
                              void* d_out, int out_size, void* d_ws, size_t ws_size,
                              hipStream_t stream) {
  (void)in_sizes; (void)n_in; (void)out_size; (void)ws_size;
  const int*   z     = (const int*)d_in[0];
  const float* pos   = (const float*)d_in[1];
  const int*   eidx  = (const int*)d_in[2];
  const int*   batch = (const int*)d_in[3];
  const float* emb   = (const float*)d_in[4];
  const float* eW1   = (const float*)d_in[5];
  const float* eb1   = (const float*)d_in[6];
  const float* eW2   = (const float*)d_in[7];
  const float* eb2   = (const float*)d_in[8];
  const float* cW    = (const float*)d_in[9];
  const float* cb    = (const float*)d_in[10];
  const float* nW1   = (const float*)d_in[11];
  const float* nb1   = (const float*)d_in[12];
  const float* nW2   = (const float*)d_in[13];
  const float* nb2   = (const float*)d_in[14];
  const float* qW1   = (const float*)d_in[15];
  const float* qb1   = (const float*)d_in[16];
  const float* qW2   = (const float*)d_in[17];
  const float* qb2   = (const float*)d_in[18];
  float* mu = (float*)d_out;

  char* p = (char*)d_ws;
  __bf16* hb    = (__bf16*)p; p += (size_t)NN * NF * 2;
  float* x      = (float*)p;  p += (size_t)NN * 3 * 4;
  float* m_agg  = (float*)p;  p += (size_t)NN * NF * 4;
  float* dxb    = (float*)p;  p += (size_t)NN * 3 * 4;
  float* q      = (float*)p;  p += (size_t)NN * 4;
  float* cnt    = (float*)p;  p += (size_t)NB * 4;
  float* xsum   = (float*)p;  p += (size_t)NB * 3 * 4;
  float* center = (float*)p;  p += (size_t)NB * 3 * 4;
  __bf16* eW1t  = (__bf16*)p; p += (size_t)NLAY * 64 * W1T_STRIDE * 2;
  __bf16* eW2t  = (__bf16*)p; p += (size_t)NLAY * 64 * 64 * 2;
  __bf16* nW1t  = (__bf16*)p; p += (size_t)NLAY * 64 * 128 * 2;
  __bf16* nW2t  = (__bf16*)p; p += (size_t)NLAY * 64 * 64 * 2;
  __bf16* qW1t  = (__bf16*)p; p += (size_t)64 * 64 * 2;

  const int* erow = eidx;
  const int* ecol = eidx + NE;

  k_prep<<<(NLAY * EDGE_IN * 64 + 255) / 256, 256, 0, stream>>>(
      eW1, eW2, nW1, nW2, qW1, eW1t, eW2t, nW1t, nW2t, qW1t);
  k_init<<<(NN * NF + 255) / 256, 256, 0, stream>>>(z, pos, emb, hb, x, m_agg, dxb, cnt, xsum);

  const int edgeBlocks = NE / 128;            // 12500, exact
  const int nodeBlocks = (NN / 16 + 7) / 8;   // 391

  for (int l = 0; l < NLAY; ++l) {
    k_edge<<<edgeBlocks, 256, 0, stream>>>(hb, x, m_agg, dxb, erow, ecol,
        eW1t + l * 64 * W1T_STRIDE, eb1 + l * NH,
        eW2t + l * 64 * 64,         eb2 + l * NH,
        cW + l * NH,                cb + l);
    k_node<<<nodeBlocks, 256, 0, stream>>>(hb, x, m_agg, dxb,
        nW1t + l * 64 * 128, nb1 + l * NH,
        nW2t + l * 64 * 64,  nb2 + l * NF);
  }

  k_q<<<nodeBlocks, 256, 0, stream>>>(hb, x, batch, qW1t, qb1, qW2, qb2, q, cnt, xsum);
  k_center<<<1, NB, 0, stream>>>(cnt, xsum, center, mu);
  k_mu<<<(NN + 255) / 256, 256, 0, stream>>>(x, q, batch, center, mu);
}
